// EncoderBlockWithDCNv3_79860621902186
// MI455X (gfx1250) — matrix operand order repaired and checked by reading, not compiled, not run on hardware
//
#include <hip/hip_runtime.h>
#include <math.h>

#define DD 32
#define HH 64
#define WW 64
#define NTOT (DD * HH * WW)   // 131072 spatial points
#define NSLAB 16384           // column slab for offset/mask buffers (divides NTOT)
#define KPTS 27
#define TILE_N 256
#define LDSP 260              // 256 + 4 pad dwords per row (TDM pad feature)

typedef __attribute__((ext_vector_type(16))) __bf16 v16bf;
typedef __attribute__((ext_vector_type(8)))  float  v8f;
typedef __attribute__((ext_vector_type(4)))  unsigned int u32x4;
typedef __attribute__((ext_vector_type(8)))  int    i32x8;
typedef __attribute__((ext_vector_type(4)))  int    i32x4;

__device__ __forceinline__ float gelu_f(float x) {
  return 0.5f * x * (1.0f + erff(x * 0.7071067811865475f));
}

// ---------------------------------------------------------------------------
// TDM: DMA a [32 rows x 256 cols] fp32 tile (global row stride = pitch elems)
// into LDS, padding +4 dwords after every 256-dword row (bank rotation).
// D# layout per CDNA5 ISA 8.3/8.4: group0 = {count/lds_addr/global_addr/type},
// group1 = {flags, tensor dims, tile dims, dim0 stride}. Groups 2/3 zero (2D).
// 6-arg builtin form (clang-23 / therock-10.0 headers).
// ---------------------------------------------------------------------------
__device__ __forceinline__ void tdm_load_tile(unsigned lds_byte_off,
                                              const float* gsrc,
                                              unsigned long long pitch_elems) {
  unsigned long long ga = (unsigned long long)(uintptr_t)gsrc;
  u32x4 g0;
  g0[0] = 1u;                                               // count=1, user mode
  g0[1] = lds_byte_off;                                     // lds_addr (bytes)
  g0[2] = (unsigned)(ga & 0xFFFFFFFFu);                     // global_addr lo
  g0[3] = (unsigned)((ga >> 32) & 0x1FFFFFFu) | (2u << 30); // addr hi | type=2
  const unsigned td0 = 256u, td1 = 32u;                     // tensor dims (tile-exact)
  const unsigned tw0 = 256u, tw1 = 32u;                     // tile dims
  i32x8 g1;
  // data_size=2 (4B) | pad_enable | pad_interval=7 (256 dw) | pad_amount=3 (4 dw)
  g1[0] = (int)((2u << 16) | (1u << 20) | (7u << 22) | (3u << 25));
  g1[1] = (int)((td0 & 0xFFFFu) << 16);
  g1[2] = (int)(((td0 >> 16) & 0xFFFFu) | ((td1 & 0xFFFFu) << 16));
  g1[3] = (int)(((td1 >> 16) & 0xFFFFu) | ((tw0 & 0xFFFFu) << 16));
  g1[4] = (int)(tw1 & 0xFFFFu);                             // tile_dim1 | tile_dim2=0
  g1[5] = (int)(pitch_elems & 0xFFFFFFFFull);               // dim0 stride lo
  g1[6] = (int)((pitch_elems >> 32) & 0xFFFFull);           // stride hi | dim1 stride=0
  g1[7] = 0;
  i32x4 gz  = {0, 0, 0, 0};
  i32x8 gz8 = {0, 0, 0, 0, 0, 0, 0, 0};
  __builtin_amdgcn_tensor_load_to_lds(g0, g1, gz, gz, gz8, 0);
}

// ---------------------------------------------------------------------------
// WMMA GEMM: Y[M, ncols] = W[M, K] x X[K, ncols-window]  (fp32 in/out, bf16 MAC)
// Block = 128 threads (4 waves); block tile = 16(M) x 256(N); wave strip 16x64.
// X tile staged to LDS by the Tensor Data Mover; A lane layout & B lane layout
// follow ISA 7.12.2 for 16-bit 16x32 A and 32x16 B.
// ---------------------------------------------------------------------------
__global__ __launch_bounds__(128) void gemm_wmma_k(
    const float* __restrict__ Wm, const float* __restrict__ X,
    float* __restrict__ Y, const float* __restrict__ bias,
    int M, int K, long long xpitch, long long xcol0, long long ypitch,
    int withBias)
{
  __shared__ float tileB[32 * LDSP];
  const int lane   = threadIdx.x & 31;
  const int wave   = threadIdx.x >> 5;
  const long long nbase = (long long)blockIdx.x * TILE_N;
  const int mbase  = blockIdx.y * 16;
  const int row    = mbase + (lane & 15);
  const int koff   = (lane < 16) ? 0 : 8;    // A: K sub-block per half-wave
  const int klocal = (lane < 16) ? 0 : 16;   // unused; B also indexes K via koff
  v8f acc[4] = {};

  for (int kb = 0; kb < K; kb += 32) {
    if (wave == 0) {
      tdm_load_tile((unsigned)(uintptr_t)tileB,
                    X + (long long)kb * xpitch + xcol0 + nbase,
                    (unsigned long long)xpitch);
      __builtin_amdgcn_s_wait_tensorcnt(0);
    }
    __syncthreads();

    // A fragment: lane<16 holds row M=lane, K {0..7,16..23}; lane>=16 same row,
    // K {8..15,24..31}.  (16-bit A 16x32 VGPR layout.)
    v16bf afrag;
#pragma unroll
    for (int e = 0; e < 16; ++e) {
      const int k = kb + koff + e + ((e >= 8) ? 8 : 0);
      const float wv = (row < M) ? Wm[(long long)row * K + k] : 0.0f;
      afrag[e] = (__bf16)wv;
    }
#pragma unroll
    for (int nt = 0; nt < 4; ++nt) {
      const int ncol = wave * 64 + nt * 16 + (lane & 15);
      v16bf bfrag;   // B 32x16: lanes 0-15 K {0..7,16..23}, lanes 16-31 K {8..15,24..31}
#pragma unroll
      for (int e = 0; e < 16; ++e)
        bfrag[e] = (__bf16)tileB[(koff + e + ((e >= 8) ? 8 : 0)) * LDSP + ncol];
      acc[nt] = __builtin_amdgcn_wmma_f32_16x16x32_bf16(
          false, afrag, false, bfrag, (short)0, acc[nt], false, false);
    }
    __syncthreads();
  }

  // C/D layout: VGPR r -> M = mbase + r + (lane<16 ? 0 : 8), N = lane&15.
#pragma unroll
  for (int nt = 0; nt < 4; ++nt) {
    const long long col = nbase + wave * 64 + nt * 16 + (lane & 15);
#pragma unroll
    for (int r = 0; r < 8; ++r) {
      const int ro = mbase + r + ((lane >> 4) << 3);
      if (ro < M) {
        float yv = acc[nt][r];
        if (withBias) yv += bias[ro];
        Y[(long long)ro * ypitch + col] = yv;
      }
    }
  }
}

// ---------------------------------------------------------------------------
// Depthwise 3x3x3 conv (pad 1) + exact GELU.  v,z: [C][32][64][64], w: [C][27]
// ---------------------------------------------------------------------------
__global__ void dwconv3_gelu_k(const float* __restrict__ v,
                               const float* __restrict__ wdw,
                               float* __restrict__ z, int C)
{
  long long i = (long long)blockIdx.x * blockDim.x + threadIdx.x;
  if (i >= (long long)C * NTOT) return;
  const int c = (int)(i >> 17);          // NTOT = 2^17
  const int s = (int)(i & (NTOT - 1));
  const int d = s >> 12, h = (s >> 6) & 63, w = s & 63;
  const float* vp = v + (long long)c * NTOT;
  const float* wp = wdw + c * 27;
  float a = 0.0f;
#pragma unroll
  for (int kd = 0; kd < 3; ++kd)
#pragma unroll
    for (int kh = 0; kh < 3; ++kh)
#pragma unroll
      for (int kw = 0; kw < 3; ++kw) {
        const int dd = d + kd - 1, hh = h + kh - 1, ww = w + kw - 1;
        if (dd >= 0 && dd < DD && hh >= 0 && hh < HH && ww >= 0 && ww < WW)
          a += wp[(kd * 3 + kh) * 3 + kw] * vp[((long long)dd << 12) + (hh << 6) + ww];
      }
  z[i] = gelu_f(a);
}

// ---------------------------------------------------------------------------
// Softmax over the 27 sampling points, in place.  m: [G*27][ns]
// ---------------------------------------------------------------------------
__global__ void softmax27_k(float* __restrict__ m, int G, int ns)
{
  const int t = blockIdx.x * blockDim.x + threadIdx.x;
  if (t >= G * ns) return;
  const int g = t / ns, j = t - g * ns;
  float* p = m + (long long)g * KPTS * ns + j;
  float mx = -3.4e38f;
  for (int k = 0; k < KPTS; ++k) mx = fmaxf(mx, p[(long long)k * ns]);
  float sum = 0.0f;
  for (int k = 0; k < KPTS; ++k) {
    const float e = expf(p[(long long)k * ns] - mx);
    p[(long long)k * ns] = e;
    sum += e;
  }
  const float r = 1.0f / sum;
  for (int k = 0; k < KPTS; ++k) p[(long long)k * ns] *= r;
}

// ---------------------------------------------------------------------------
// Deformable trilinear aggregation.  One thread per (group, point); Cg-channel
// accumulator in registers.  v/acc: [G*CG][NTOT], off: [G*81][ns], msk: [G*27][ns]
// ---------------------------------------------------------------------------
template <int CG>
__global__ void deform_k(const float* __restrict__ v, const float* __restrict__ off,
                         const float* __restrict__ msk, float* __restrict__ acc,
                         int G, int nbase, int ns, float sd, float sh, float sw)
{
  const int t = blockIdx.x * blockDim.x + threadIdx.x;
  if (t >= G * ns) return;
  const int g = t / ns, j = t - g * ns;
  const int n = nbase + j;
  const int d = n >> 12, h = (n >> 6) & 63, w = n & 63;
  const float* vg = v + (long long)g * CG * NTOT;
  float a[CG];
#pragma unroll
  for (int c = 0; c < CG; ++c) a[c] = 0.0f;

  for (int k = 0; k < KPTS; ++k) {
    const int kd = k / 9, rr = k - kd * 9, kh = rr / 3, kw = rr - kh * 3;
    const long long ob = ((long long)(g * 81 + k * 3)) * ns + j;
    const float pd = (float)(d + kd - 1) + off[ob] * sd;
    const float ph = (float)(h + kh - 1) + off[ob + ns] * sh;
    const float pw = (float)(w + kw - 1) + off[ob + 2LL * ns] * sw;
    const float mk = msk[((long long)(g * KPTS + k)) * ns + j];
    const int z0 = (int)floorf(pd), y0 = (int)floorf(ph), x0 = (int)floorf(pw);
    const float fz = pd - (float)z0, fy = ph - (float)y0, fx = pw - (float)x0;
#pragma unroll
    for (int dz = 0; dz < 2; ++dz) {
      const int iz = z0 + dz;
      if (iz < 0 || iz >= DD) continue;
      const float wz = dz ? fz : 1.0f - fz;
#pragma unroll
      for (int dy = 0; dy < 2; ++dy) {
        const int iy = y0 + dy;
        if (iy < 0 || iy >= HH) continue;
        const float wy = dy ? fy : 1.0f - fy;
#pragma unroll
        for (int dx = 0; dx < 2; ++dx) {
          const int ix = x0 + dx;
          if (ix < 0 || ix >= WW) continue;
          const float wt = mk * wz * wy * (dx ? fx : 1.0f - fx);
          const float* vp = vg + (((long long)iz << 12) + (iy << 6) + ix);
#pragma unroll
          for (int c = 0; c < CG; ++c) a[c] += wt * vp[(long long)c * NTOT];
        }
      }
    }
  }
#pragma unroll
  for (int c = 0; c < CG; ++c)
    acc[((long long)(g * CG + c)) * NTOT + n] = a[c];
}

// ---------------------------------------------------------------------------
// InstanceNorm (affine=False, eps 1e-5) + GELU, in place; one block per channel.
// ---------------------------------------------------------------------------
__global__ void inorm_gelu_k(float* __restrict__ y)
{
  __shared__ float s1[256], s2[256];
  const int c = blockIdx.x, t = threadIdx.x;
  float* p = y + (long long)c * NTOT;
  float sum = 0.0f, sq = 0.0f;
  for (int i = t; i < NTOT; i += 256) { const float x = p[i]; sum += x; sq += x * x; }
  s1[t] = sum; s2[t] = sq;
  __syncthreads();
  for (int s = 128; s > 0; s >>= 1) {
    if (t < s) { s1[t] += s1[t + s]; s2[t] += s2[t + s]; }
    __syncthreads();
  }
  const float mu  = s1[0] * (1.0f / NTOT);
  const float var = s2[0] * (1.0f / NTOT) - mu * mu;
  const float rs  = rsqrtf(var + 1e-5f);
  for (int i = t; i < NTOT; i += 256) p[i] = gelu_f((p[i] - mu) * rs);
}

__global__ void residual_k(const float* __restrict__ c, const float* __restrict__ r,
                           const float* __restrict__ gate, float* __restrict__ out)
{
  long long i = (long long)blockIdx.x * blockDim.x + threadIdx.x;
  if (i >= 64LL * NTOT) return;
  const float s = 1.0f / (1.0f + expf(-gate[0]));
  out[i] = c[i] + s * r[i];
}

__global__ void maxpool_k(const float* __restrict__ x, float* __restrict__ o)
{
  const long long total = 64LL * 16 * 32 * 32;
  long long i = (long long)blockIdx.x * blockDim.x + threadIdx.x;
  if (i >= total) return;
  const int c  = (int)(i >> 14);
  const int rm = (int)(i & 16383);
  const int od = rm >> 10, oh = (rm >> 5) & 31, ow = rm & 31;
  const float* b = x + ((long long)c << 17) + ((long long)(od * 2) << 12)
                     + ((oh * 2) << 6) + ow * 2;
  float m = -3.4e38f;
#pragma unroll
  for (int dz = 0; dz < 2; ++dz)
#pragma unroll
    for (int dy = 0; dy < 2; ++dy)
#pragma unroll
      for (int dx = 0; dx < 2; ++dx)
        m = fmaxf(m, b[((long long)dz << 12) + (dy << 6) + dx]);
  o[i] = m;
}

// ---------------------------------------------------------------------------
// Host-side orchestration
// ---------------------------------------------------------------------------
static inline long long cdivll(long long a, long long b) { return (a + b - 1) / b; }

static void launch_gemm(hipStream_t st, const float* Wm, const float* X,
                        long long xpitch, long long xcol0, float* Y, long long ypitch,
                        const float* bias, int M, int K, long long ncols)
{
  dim3 grid((unsigned)(ncols / TILE_N), (unsigned)cdivll(M, 16));
  gemm_wmma_k<<<grid, dim3(128), 0, st>>>(Wm, X, Y, bias, M, K,
                                          xpitch, xcol0, ypitch, bias ? 1 : 0);
}

static void run_dcn(hipStream_t st, const float* in, int Cin, int Cout, int G,
                    const float* win, const float* wdw, const float* woff,
                    const float* boff, const float* wmask, const float* bmask,
                    const float* wout, float sd, float sh, float sw,
                    float* v, float* z, float* acc, float* out,
                    float* offb, float* mskb)
{
  // value projection: v = win @ in
  launch_gemm(st, win, in, NTOT, 0, v, NTOT, nullptr, Cin, Cin, NTOT);
  { const long long tot = (long long)Cin * NTOT;
    dwconv3_gelu_k<<<(unsigned)cdivll(tot, 256), 256, 0, st>>>(v, wdw, z, Cin); }
  const int Cg = Cin / G;
  for (int nb = 0; nb < NTOT; nb += NSLAB) {
    launch_gemm(st, woff,  z, NTOT, nb, offb, NSLAB, boff,  G * KPTS * 3, Cin, NSLAB);
    launch_gemm(st, wmask, z, NTOT, nb, mskb, NSLAB, bmask, G * KPTS,     Cin, NSLAB);
    const int t1 = G * NSLAB;
    softmax27_k<<<(unsigned)cdivll(t1, 256), 256, 0, st>>>(mskb, G, NSLAB);
    if (Cg == 8)
      deform_k<8><<<(unsigned)cdivll(t1, 256), 256, 0, st>>>(v, offb, mskb, acc, G, nb, NSLAB, sd, sh, sw);
    else if (Cg == 16)
      deform_k<16><<<(unsigned)cdivll(t1, 256), 256, 0, st>>>(v, offb, mskb, acc, G, nb, NSLAB, sd, sh, sw);
    else
      deform_k<32><<<(unsigned)cdivll(t1, 256), 256, 0, st>>>(v, offb, mskb, acc, G, nb, NSLAB, sd, sh, sw);
  }
  // output projection + instance-norm + GELU
  launch_gemm(st, wout, acc, NTOT, 0, out, NTOT, nullptr, Cout, Cin, NTOT);
  inorm_gelu_k<<<Cout, 256, 0, st>>>(out);
}

extern "C" void kernel_launch(void* const* d_in, const int* in_sizes, int n_in,
                              void* d_out, int out_size, void* d_ws, size_t ws_size,
                              hipStream_t stream)
{
  (void)in_sizes; (void)n_in; (void)out_size; (void)ws_size;
  const float* x = (const float*)d_in[0];
  const float *d1w[7], *d2w[7], *d3w[7];
  for (int i = 0; i < 7; ++i) { d1w[i] = (const float*)d_in[1 + i];
                                d2w[i] = (const float*)d_in[8 + i];
                                d3w[i] = (const float*)d_in[16 + i]; }
  const float* pre_w  = (const float*)d_in[15];
  const float* post_w = (const float*)d_in[23];
  const float* gate   = (const float*)d_in[24];

  float* ws = (float*)d_ws;
  const long long NF = 64LL * NTOT;
  float* bufA = ws;            // layer outputs / r0 / r1
  float* bufB = ws + NF;       // c2 (kept for the residual)
  float* bufV = ws + 2 * NF;   // value tensor / r2
  float* bufZ = ws + 3 * NF;   // offset-branch features
  float* bufC = ws + 4 * NF;   // deformable accumulator
  float* offb = ws + 5 * NF;                      // 324 * NSLAB
  float* mskb = offb + 324LL * NSLAB;             // 108 * NSLAB

  // d1: 32 -> 64, G=4
  run_dcn(stream, x, 32, 64, 4, d1w[0], d1w[1], d1w[2], d1w[3], d1w[4], d1w[5], d1w[6],
          1.0f, 1.0f, 1.0f, bufV, bufZ, bufC, bufA, offb, mskb);
  // d2: 64 -> 64, G=4   (c = bufB)
  run_dcn(stream, bufA, 64, 64, 4, d2w[0], d2w[1], d2w[2], d2w[3], d2w[4], d2w[5], d2w[6],
          1.0f, 1.0f, 1.0f, bufV, bufZ, bufC, bufB, offb, mskb);
  // r0 = pre_w @ c -> bufA
  launch_gemm(stream, pre_w, bufB, NTOT, 0, bufA, NTOT, nullptr, 64, 64, NTOT);
  // d3: 64 -> 64, G=2, axis_scale = (0.5, 1, 1); in-place safe (input only read first)
  run_dcn(stream, bufA, 64, 64, 2, d3w[0], d3w[1], d3w[2], d3w[3], d3w[4], d3w[5], d3w[6],
          0.5f, 1.0f, 1.0f, bufV, bufZ, bufC, bufA, offb, mskb);
  // r2 = post_w @ r1 -> bufV
  launch_gemm(stream, post_w, bufA, NTOT, 0, bufV, NTOT, nullptr, 64, 64, NTOT);
  // conv_out = c + sigmoid(gate) * r2  -> d_out[0 .. 64*NTOT)
  float* outp = (float*)d_out;
  residual_k<<<(unsigned)cdivll(64LL * NTOT, 256), 256, 0, stream>>>(bufB, bufV, gate, outp);
  // pool_out -> d_out[64*NTOT ..)
  maxpool_k<<<(unsigned)cdivll(64LL * 16 * 32 * 32, 256), 256, 0, stream>>>(outp, outp + NF);
}
